// MultiLoRALinear_81638738362865
// MI455X (gfx1250) — matrix operand order repaired and checked by reading, not compiled, not run on hardware
//
#include <hip/hip_runtime.h>
#include <stdint.h>

// Problem constants (from reference)
#define T_TOK   16384
#define IN_DIM  4096
#define OUT_DIM 4096
#define NADAPT  8
#define RANK    16

// GEMM tiling
#define BM 128
#define BN 128
#define BK 32
#define LDK 40   // padded LDS row stride (halves) to spread banks

typedef __attribute__((ext_vector_type(8)))  float          v8f;
typedef __attribute__((ext_vector_type(16))) __bf16         v16bf;
typedef __attribute__((ext_vector_type(4)))  unsigned short v4us;
typedef __attribute__((ext_vector_type(8)))  unsigned short v8us;
typedef __attribute__((ext_vector_type(16))) unsigned short v16us;

__device__ __forceinline__ unsigned short rne_bf16_bits(float f) {
  unsigned u = __float_as_uint(f);
  u += 0x7fffu + ((u >> 16) & 1u);
  return (unsigned short)(u >> 16);
}

// Truncation-hi split: hi = top 16 bits (exact value hf), lo = RNE(f - hf).
// f - hf is exact, so total representation error ~2^-17 relative.
__device__ __forceinline__ void split_bf16(float f, unsigned short& hi,
                                           unsigned short& lo) {
  unsigned u = __float_as_uint(f);
  hi = (unsigned short)(u >> 16);
  float hf = __uint_as_float(u & 0xffff0000u);
  lo = rne_bf16_bits(f - hf);
}

// Build a v16bf WMMA operand from two 16-byte LDS chunks (ds_load_b128 each).
__device__ __forceinline__ v16bf ld_frag(const unsigned short* p0,
                                         const unsigned short* p1) {
  v8us a = *(const v8us*)p0;
  v8us b = *(const v8us*)p1;
  v16us t;
#pragma unroll
  for (int i = 0; i < 8; ++i) { t[i] = a[i]; t[i + 8] = b[i]; }
  return __builtin_bit_cast(v16bf, t);
}

// CDNA5 async copy: global (bf16, 16B/lane) -> LDS, tracked by ASYNCcnt.
// Low 32 bits of a generic pointer to __shared__ are the LDS segment offset.
__device__ __forceinline__ void async_load_b128(void* lds_ptr,
                                                const unsigned short* base,
                                                unsigned byte_off) {
  unsigned lds_off = (unsigned)(uintptr_t)lds_ptr;
  asm volatile("global_load_async_to_lds_b128 %0, %1, %2"
               :: "v"(lds_off), "v"(byte_off), "s"(base)
               : "memory");
}
__device__ __forceinline__ void wait_asynccnt0() {
  asm volatile("s_wait_asynccnt 0x0" ::: "memory");
}

// ---------------------------------------------------------------------------
// Kernel 1: per-token adapter id (searchsorted) + LoRA down-projection
//   xa[t, r] = sum_k x[t,k] * lora_A[seg(t), k, r]      (r = 0..15)
// ---------------------------------------------------------------------------
__global__ void __launch_bounds__(256)
lora_down_kernel(const float* __restrict__ x,
                 const float* __restrict__ lora_A,
                 const int*   __restrict__ offsets,
                 int*         __restrict__ seg_ws,
                 float*       __restrict__ xa_ws) {
  __shared__ float xs[IN_DIM];
  __shared__ float red[256];

  const int t   = blockIdx.x;
  const int tid = threadIdx.x;
  const float* xrow = x + (size_t)t * IN_DIM;

  for (int i = tid; i < IN_DIM; i += 256) xs[i] = xrow[i];

  int seg = 0;  // seg = #offsets <= t  (searchsorted right)
#pragma unroll
  for (int i = 0; i < NADAPT; ++i) seg += (offsets[i] <= t) ? 1 : 0;

  __syncthreads();

  const int grp = tid >> 4;
  const int r   = tid & 15;
  const float* A = lora_A + (size_t)seg * IN_DIM * RANK;
  const int k0 = grp * (IN_DIM / 16);
  float acc = 0.f;
  for (int k = 0; k < IN_DIM / 16; ++k)
    acc += xs[k0 + k] * A[(size_t)(k0 + k) * RANK + r];

  red[tid] = acc;
  __syncthreads();
  if (tid < RANK) {
    float s = 0.f;
#pragma unroll
    for (int g = 0; g < 16; ++g) s += red[g * 16 + tid];
    xa_ws[(size_t)t * RANK + tid] = s;
  }
  if (tid == 0) seg_ws[t] = seg;
}

// ---------------------------------------------------------------------------
// Pre-split an f32 array into bf16 hi / lo arrays (one pass, HBM-bound).
// ---------------------------------------------------------------------------
__global__ void __launch_bounds__(256)
split_bf16_kernel(const float* __restrict__ src,
                  unsigned short* __restrict__ hi,
                  unsigned short* __restrict__ lo, int n4) {
  int i = blockIdx.x * 256 + threadIdx.x;
  if (i >= n4) return;
  float4 v = *(const float4*)(src + (size_t)i * 4);
  float f[4] = {v.x, v.y, v.z, v.w};
  v4us h, l;
#pragma unroll
  for (int j = 0; j < 4; ++j) {
    unsigned short hb, lb;
    split_bf16(f[j], hb, lb);
    h[j] = hb; l[j] = lb;
  }
  *(v4us*)(hi + (size_t)i * 4) = h;
  *(v4us*)(lo + (size_t)i * 4) = l;
}

// ---------------------------------------------------------------------------
// Kernel 2: out = x @ W^T + b + xa @ lora_B[seg]
// bf16 hi/lo split-precision GEMM on v_wmma_f32_16x16x32_bf16.
// PREA/PREB: operand pre-split in global memory -> async-copy straight to LDS.
// ---------------------------------------------------------------------------
template<bool PREA, bool PREB>
__global__ void __launch_bounds__(256)
gemm_lora_kernel(const float* __restrict__ x,
                 const float* __restrict__ W,
                 const float* __restrict__ bias,
                 const float* __restrict__ lora_B,
                 const int*   __restrict__ seg_ws,
                 const float* __restrict__ xa_ws,
                 const unsigned short* __restrict__ xhi_g,
                 const unsigned short* __restrict__ xlo_g,
                 const unsigned short* __restrict__ whi_g,
                 const unsigned short* __restrict__ wlo_g,
                 float* __restrict__ out) {
  __shared__ unsigned short Ahi[BM][LDK];
  __shared__ unsigned short Alo[BM][LDK];
  __shared__ unsigned short Bhi[BN][LDK];
  __shared__ unsigned short Blo[BN][LDK];

  const int tid  = threadIdx.x;
  const int gm   = blockIdx.y * BM;
  const int gn   = blockIdx.x * BN;

  const int wave = tid >> 5;
  const int lane = tid & 31;
  const int half = lane >> 4;   // wave32: lanes 0-15 / 16-31
  const int ln   = lane & 15;
  const int wm   = (wave & 1) * 64;   // 2 waves along M
  const int wn   = (wave >> 1) * 32;  // 4 waves along N

  v8f acc[4][2];
#pragma unroll
  for (int mt = 0; mt < 4; ++mt)
#pragma unroll
    for (int nt = 0; nt < 2; ++nt) acc[mt][nt] = 0.0f;

  const int ksteps = IN_DIM / BK;
  for (int ks = 0; ks < ksteps; ++ks) {
    const int kb = ks * BK;

    // ---- stage (fallback path): global f32 loads before the barrier ----
    float4 av[4], bv[4];
    if (!PREA) {
#pragma unroll
      for (int i = 0; i < 4; ++i) {
        const int idx = tid + i * 256;
        const int row = idx >> 3;
        const int kv  = (idx & 7) * 4;
        av[i] = *(const float4*)(x + (size_t)(gm + row) * IN_DIM + kb + kv);
      }
    }
    if (!PREB) {
#pragma unroll
      for (int i = 0; i < 4; ++i) {
        const int idx = tid + i * 256;
        const int row = idx >> 3;
        const int kv  = (idx & 7) * 4;
        bv[i] = *(const float4*)(W + (size_t)(gn + row) * IN_DIM + kb + kv);
      }
    }
    if (!PREA && !PREB && ks + 1 < ksteps) {  // L2 prefetch of next K tile
      const int row = tid >> 3;
      const int kv  = (tid & 7) * 4;
      __builtin_prefetch(x + (size_t)(gm + row) * IN_DIM + kb + BK + kv, 0, 3);
      __builtin_prefetch(W + (size_t)(gn + row) * IN_DIM + kb + BK + kv, 0, 3);
    }

    __syncthreads();  // previous iteration's fragment reads are done

    // ---- fill LDS tiles ----
    if (PREA) {
#pragma unroll
      for (int i = 0; i < 2; ++i) {
        const int idx = tid + i * 256;   // 0..511
        const int row = idx >> 2;        // 0..127
        const int s   = idx & 3;         // 16B chunk within 64B row
        const unsigned goff =
            (unsigned)((((gm + row) * IN_DIM) + kb + s * 8) * 2);
        async_load_b128(&Ahi[row][s * 8], xhi_g, goff);
        async_load_b128(&Alo[row][s * 8], xlo_g, goff);
      }
    } else {
#pragma unroll
      for (int i = 0; i < 4; ++i) {
        const int idx = tid + i * 256;
        const int row = idx >> 3;
        const int kv  = (idx & 7) * 4;
        const float fa[4] = {av[i].x, av[i].y, av[i].z, av[i].w};
#pragma unroll
        for (int j = 0; j < 4; ++j) {
          unsigned short hb, lb;
          split_bf16(fa[j], hb, lb);
          Ahi[row][kv + j] = hb;
          Alo[row][kv + j] = lb;
        }
      }
    }
    if (PREB) {
#pragma unroll
      for (int i = 0; i < 2; ++i) {
        const int idx = tid + i * 256;
        const int row = idx >> 2;
        const int s   = idx & 3;
        const unsigned goff =
            (unsigned)((((gn + row) * IN_DIM) + kb + s * 8) * 2);
        async_load_b128(&Bhi[row][s * 8], whi_g, goff);
        async_load_b128(&Blo[row][s * 8], wlo_g, goff);
      }
    } else {
#pragma unroll
      for (int i = 0; i < 4; ++i) {
        const int idx = tid + i * 256;
        const int row = idx >> 3;
        const int kv  = (idx & 7) * 4;
        const float fb[4] = {bv[i].x, bv[i].y, bv[i].z, bv[i].w};
#pragma unroll
        for (int j = 0; j < 4; ++j) {
          unsigned short hb, lb;
          split_bf16(fb[j], hb, lb);
          Bhi[row][kv + j] = hb;
          Blo[row][kv + j] = lb;
        }
      }
    }
    if (PREA || PREB) wait_asynccnt0();  // own async writes landed
    __syncthreads();

    // ---- fragment loads (16-bit A 16x32 / B 32x16 VGPR layouts) ----
    v16bf ah[4], al[4], bh[2], bl[2];
#pragma unroll
    for (int mt = 0; mt < 4; ++mt) {
      const int r = wm + mt * 16 + ln;
      ah[mt] = ld_frag(&Ahi[r][half * 8], &Ahi[r][16 + half * 8]);
      al[mt] = ld_frag(&Alo[r][half * 8], &Alo[r][16 + half * 8]);
    }
#pragma unroll
    for (int nt = 0; nt < 2; ++nt) {
      const int r = wn + nt * 16 + ln;
      bh[nt] = ld_frag(&Bhi[r][half * 8], &Bhi[r][16 + half * 8]);
      bl[nt] = ld_frag(&Blo[r][half * 8], &Blo[r][16 + half * 8]);
    }

    // ---- 24 WMMAs: hi*hi + hi*lo + lo*hi (f32 accumulate) ----
#pragma unroll
    for (int mt = 0; mt < 4; ++mt) {
#pragma unroll
      for (int nt = 0; nt < 2; ++nt) {
        acc[mt][nt] = __builtin_amdgcn_wmma_f32_16x16x32_bf16(
            false, ah[mt], false, bh[nt], (short)0, acc[mt][nt], false, false);
        acc[mt][nt] = __builtin_amdgcn_wmma_f32_16x16x32_bf16(
            false, ah[mt], false, bl[nt], (short)0, acc[mt][nt], false, false);
        acc[mt][nt] = __builtin_amdgcn_wmma_f32_16x16x32_bf16(
            false, al[mt], false, bh[nt], (short)0, acc[mt][nt], false, false);
      }
    }
  }

  // ---- epilogue: + bias + rank-16 LoRA up-projection, fused writeback ----
#pragma unroll
  for (int nt = 0; nt < 2; ++nt) {
    const int col = gn + wn + nt * 16 + ln;
    const float bs = bias[col];
#pragma unroll
    for (int mt = 0; mt < 4; ++mt) {
      const int rbase = gm + wm + mt * 16 + half * 8;
      int segc = -1;
      float lb[RANK];
#pragma unroll
      for (int j = 0; j < 8; ++j) {
        const int rowg = rbase + j;
        const int seg  = seg_ws[rowg];
        if (seg != segc) {  // seg is piecewise-constant: rare reload
          segc = seg;
#pragma unroll
          for (int r = 0; r < RANK; ++r)
            lb[r] = lora_B[((size_t)seg * RANK + r) * OUT_DIM + col];
        }
        const float* xa = xa_ws + (size_t)rowg * RANK;
        float ls = 0.f;
#pragma unroll
        for (int r = 0; r < RANK; ++r) ls += xa[r] * lb[r];
        out[(size_t)rowg * OUT_DIM + col] = acc[mt][nt][j] + bs + ls;
      }
    }
  }
}

// ---------------------------------------------------------------------------
extern "C" void kernel_launch(void* const* d_in, const int* in_sizes, int n_in,
                              void* d_out, int out_size, void* d_ws, size_t ws_size,
                              hipStream_t stream) {
  (void)in_sizes; (void)n_in; (void)out_size;
  const float* x       = (const float*)d_in[0];
  const float* W       = (const float*)d_in[1];
  const float* b       = (const float*)d_in[2];
  const float* lora_A  = (const float*)d_in[3];
  const float* lora_B  = (const float*)d_in[4];
  const int*   offsets = (const int*)d_in[5];
  float*       out     = (float*)d_out;

  // Workspace layout (tiered by ws_size, deterministic each call):
  //   [T] int32 seg | [T x 16] f32 xa | (align) | W hi | W lo | x hi | x lo
  size_t p = 0;
  int*   seg_ws = (int*)((char*)d_ws + p);   p += (size_t)T_TOK * 4;
  float* xa_ws  = (float*)((char*)d_ws + p); p += (size_t)T_TOK * RANK * 4;
  p = (p + 255) & ~(size_t)255;
  const size_t wsz = (size_t)OUT_DIM * IN_DIM * 2;  // 32 MB each
  unsigned short* whi = (unsigned short*)((char*)d_ws + p); p += wsz;
  unsigned short* wlo = (unsigned short*)((char*)d_ws + p); p += wsz;
  const size_t need_w = p;
  const size_t xsz = (size_t)T_TOK * IN_DIM * 2;    // 128 MB each
  unsigned short* xhi = (unsigned short*)((char*)d_ws + p); p += xsz;
  unsigned short* xlo = (unsigned short*)((char*)d_ws + p); p += xsz;
  const size_t need_x = p;

  const bool preB = ws_size >= need_w;
  const bool preA = ws_size >= need_x;

  lora_down_kernel<<<T_TOK, 256, 0, stream>>>(x, lora_A, offsets, seg_ws, xa_ws);

  if (preB) {
    const int n4 = OUT_DIM * IN_DIM / 4;
    split_bf16_kernel<<<(n4 + 255) / 256, 256, 0, stream>>>(W, whi, wlo, n4);
  }
  if (preA) {
    const int n4 = T_TOK * IN_DIM / 4;
    split_bf16_kernel<<<(n4 + 255) / 256, 256, 0, stream>>>(x, xhi, xlo, n4);
  }

  dim3 grid(OUT_DIM / BN, T_TOK / BM);
  if (preA && preB)
    gemm_lora_kernel<true, true><<<grid, dim3(256), 0, stream>>>(
        x, W, b, lora_B, seg_ws, xa_ws, xhi, xlo, whi, wlo, out);
  else if (preB)
    gemm_lora_kernel<false, true><<<grid, dim3(256), 0, stream>>>(
        x, W, b, lora_B, seg_ws, xa_ws, xhi, xlo, whi, wlo, out);
  else
    gemm_lora_kernel<false, false><<<grid, dim3(256), 0, stream>>>(
        x, W, b, lora_B, seg_ws, xa_ws, xhi, xlo, whi, wlo, out);
}
